// MS_MSA3D_13108240187574
// MI455X (gfx1250) — matrix operand order repaired and checked by reading, not compiled, not run on hardware
//
#include <hip/hip_runtime.h>
#include <math.h>

typedef __attribute__((ext_vector_type(16))) _Float16 v16h;
typedef __attribute__((ext_vector_type(8)))  _Float16 v8h;
typedef __attribute__((ext_vector_type(8)))  float    v8f;
typedef __attribute__((ext_vector_type(4)))  unsigned int u32x4;
typedef __attribute__((ext_vector_type(8)))  int      i32x8;
typedef __attribute__((ext_vector_type(4)))  int      i32x4;

#define NTOK   131072      // 8*128*128
#define CDIM   64
#define INNER  256         // 4 heads * 64
#define QKVW   768         // 3*INNER
#define HEADS  4
#define DHEAD  64

// ---------------------------------------------------------------------------
// WMMA fragment loaders (wave32); A and B both use contiguous k = 16*(lane/16) + s.
// A (16x32 f16): lanes 0-15 hold M=lane, K=0..15 in element order;
//                lanes 16-31 hold M=lane-16, K=16..31.
// B (32x16 f16): lanes 0-15 hold col n=lane, K=0..15; lanes 16-31 K=16..31.
// C/D (16x16 f32): VGPR i -> row M = i + (lane<16 ? 0 : 8), col N = lane&15.
// All f16 fragment bases below are 16B-aligned -> vector b128 / ds_b128 ops.
// ---------------------------------------------------------------------------
__device__ inline v16h frag_a_f32(const float* __restrict__ base, int ld) {
  const int lane = threadIdx.x & 31;
  const int m    = lane & 15;
  const int klo  = (lane & 16) ? 16 : 0;
  v16h a;
#pragma unroll
  for (int j = 0; j < 8; ++j) {
    a[j]     = (_Float16)base[m * ld + klo + j];
    a[8 + j] = (_Float16)base[m * ld + klo + 8 + j];
  }
  return a;
}

__device__ inline v16h frag_a_f16(const _Float16* base, int ld) {
  const int lane = threadIdx.x & 31;
  const int m    = lane & 15;
  const int klo  = (lane & 16) ? 16 : 0;
  const _Float16* p = base + m * ld + klo;
  v8h lo = *(const v8h*)(p);
  v8h hi = *(const v8h*)(p + 8);
  return __builtin_shufflevector(lo, hi, 0, 1, 2, 3, 4, 5, 6, 7,
                                 8, 9, 10, 11, 12, 13, 14, 15);
}

// B[k][n] stored row-major with stride ld (f32 source)
__device__ inline v16h frag_b_f32(const float* __restrict__ base, int ld) {
  const int lane  = threadIdx.x & 31;
  const int n     = lane & 15;
  const int kbase = (lane & 16) ? 16 : 0;
  v16h b;
#pragma unroll
  for (int j = 0; j < 16; ++j) b[j] = (_Float16)base[(kbase + j) * ld + n];
  return b;
}

// B[k][n] stored row-major with stride ld (f16 source)
__device__ inline v16h frag_b_f16(const _Float16* base, int ld) {
  const int lane  = threadIdx.x & 31;
  const int n     = lane & 15;
  const int kbase = (lane & 16) ? 16 : 0;
  v16h b;
#pragma unroll
  for (int j = 0; j < 16; ++j) b[j] = base[(kbase + j) * ld + n];
  return b;
}

// B[k][n] where memory is transposed: element (k,n) at base[n*ld + k]
// -> each lane reads 16 contiguous f16 (32B, aligned).
__device__ inline v16h frag_bT_f16(const _Float16* __restrict__ base, long ld) {
  const int lane  = threadIdx.x & 31;
  const int n     = lane & 15;
  const int kbase = (lane & 16) ? 16 : 0;
  return *(const v16h*)(base + (long)n * ld + kbase);
}

__device__ inline v8f wmma16(v16h a, v16h b, v8f c) {
  return __builtin_amdgcn_wmma_f32_16x16x32_f16(false, a, false, b, (short)0, c,
                                                false, false);
}

// ---------------------------------------------------------------------------
// Kernel 0: zero the Gram accumulator (split-K atomics need clean zeros).
// ---------------------------------------------------------------------------
__global__ void k_zero_g(float* __restrict__ G) {
  int i = blockIdx.x * blockDim.x + threadIdx.x;
  if (i < HEADS * DHEAD * DHEAD) G[i] = 0.0f;
}

// ---------------------------------------------------------------------------
// Kernel 1: QKV = X[N,64] @ Wqkv[64,768]  (WMMA f16, f32 accumulate)
// Writes q,k channel-major (for Gram contiguity), v token-major.
// One wave = one 16x16 output tile; 8 waves per block.
// ---------------------------------------------------------------------------
__global__ __launch_bounds__(256) void k_qkv(const float* __restrict__ x,
                                             const float* __restrict__ wqkv,
                                             _Float16* __restrict__ qt,
                                             _Float16* __restrict__ kt,
                                             _Float16* __restrict__ vv) {
  const long wg = (long)blockIdx.x * (blockDim.x >> 5) + (threadIdx.x >> 5);
  const int  tiles_col = QKVW / 16;                 // 48
  const long row_tile  = wg / tiles_col;
  const int  col_tile  = (int)(wg % tiles_col);
  if (row_tile >= NTOK / 16) return;                // wave-uniform
  const long row0 = row_tile * 16;
  const int  col0 = col_tile * 16;

  v8f acc = {};
#pragma unroll
  for (int k0 = 0; k0 < CDIM; k0 += 32) {
    v16h a = frag_a_f32(x + row0 * CDIM + k0, CDIM);
    v16h b = frag_b_f32(wqkv + (long)k0 * QKVW + col0, QKVW);
    acc = wmma16(a, b, acc);
  }

  const int lane  = threadIdx.x & 31;
  const int n     = lane & 15;
  const int mbase = (lane & 16) ? 8 : 0;
  const int col   = col0 + n;

  v8h pk;
#pragma unroll
  for (int i = 0; i < 8; ++i) pk[i] = (_Float16)acc[i];

  if (col < INNER) {                                 // q -> channel-major
    *(v8h*)(qt + (long)col * NTOK + row0 + mbase) = pk;   // single b128 store
  } else if (col < 2 * INNER) {                      // k -> channel-major
    *(v8h*)(kt + (long)(col - INNER) * NTOK + row0 + mbase) = pk;
  } else {                                           // v -> token-major
#pragma unroll
    for (int i = 0; i < 8; ++i)
      vv[(row0 + mbase + i) * INNER + (col - 2 * INNER)] = pk[i];
  }
}

// ---------------------------------------------------------------------------
// Kernel 2: per-channel sum of squares over N (for l2norm along tokens).
// ---------------------------------------------------------------------------
__global__ __launch_bounds__(256) void k_sumsq(const _Float16* __restrict__ qt,
                                               const _Float16* __restrict__ kt,
                                               float* __restrict__ sumsq) {
  const int ch = blockIdx.x;
  const _Float16* src =
      (ch < INNER) ? (qt + (long)ch * NTOK) : (kt + (long)(ch - INNER) * NTOK);
  float s = 0.0f;
  for (long i = threadIdx.x * 8; i < NTOK; i += (long)blockDim.x * 8) {
    v8h v = *(const v8h*)(src + i);
#pragma unroll
    for (int j = 0; j < 8; ++j) {
      float f = (float)v[j];
      s += f * f;
    }
  }
  __shared__ float red[256];
  red[threadIdx.x] = s;
  __syncthreads();
#pragma unroll
  for (int off = 128; off > 0; off >>= 1) {
    if ((int)threadIdx.x < off) red[threadIdx.x] += red[threadIdx.x + off];
    __syncthreads();
  }
  if (threadIdx.x == 0) sumsq[ch] = red[0];
}

// ---------------------------------------------------------------------------
// Kernel 3: Gram  G[h][d][e] = sum_n k_t[h,d][n] * q_t[h,e][n]
// Split-K over N, WMMA per 16x16 tile, atomicAdd into G.
// ---------------------------------------------------------------------------
__global__ __launch_bounds__(256) void k_gram(const _Float16* __restrict__ qt,
                                              const _Float16* __restrict__ kt,
                                              float* __restrict__ G, int chunk) {
  const int head = blockIdx.x;
  const int nbeg = blockIdx.y * chunk;
  const int wave = threadIdx.x >> 5;

  for (int t = wave * 2; t < wave * 2 + 2; ++t) {   // wave-uniform
    const int dt = t >> 2, et = t & 3;
    const _Float16* arow = kt + (long)(head * DHEAD + dt * 16) * NTOK;
    const _Float16* brow = qt + (long)(head * DHEAD + et * 16) * NTOK;
    v8f acc = {};
    for (int n0 = nbeg; n0 < nbeg + chunk; n0 += 32) {
      __builtin_prefetch(arow + n0 + 512, 0, 3);    // global_prefetch_b8
      __builtin_prefetch(brow + n0 + 512, 0, 3);
      v16h a = frag_a_f16(arow + n0, NTOK);
      v16h b = frag_bT_f16(brow + n0, (long)NTOK);
      acc = wmma16(a, b, acc);
    }
    const int lane = threadIdx.x & 31;
    const int e    = lane & 15;
    const int mb   = (lane & 16) ? 8 : 0;
    float* g = G + head * DHEAD * DHEAD + (dt * 16) * DHEAD + et * 16;
#pragma unroll
    for (int i = 0; i < 8; ++i) atomicAdd(&g[(mb + i) * DHEAD + e], acc[i]);
  }
}

// ---------------------------------------------------------------------------
// Kernel 4: attn = softmax_e( G[d,e] / (||k_d|| ||q_e||) * rescale[h] )
// Stored transposed f16: attn_s[h][e][d].
// ---------------------------------------------------------------------------
__global__ __launch_bounds__(64) void k_attn(const float* __restrict__ G,
                                             const float* __restrict__ sumsq,
                                             const float* __restrict__ rescale,
                                             _Float16* __restrict__ attn_s) {
  const int head = blockIdx.x;
  const int d    = threadIdx.x;
  if (d >= DHEAD) return;
  const float kn = fmaxf(sqrtf(sumsq[INNER + head * DHEAD + d]), 1e-12f);
  const float sc = rescale[head];
  float row[DHEAD];
  float mx = -INFINITY;
#pragma unroll 8
  for (int e = 0; e < DHEAD; ++e) {
    const float qn = fmaxf(sqrtf(sumsq[head * DHEAD + e]), 1e-12f);
    const float a  = G[head * DHEAD * DHEAD + d * DHEAD + e] / (kn * qn) * sc;
    row[e] = a;
    mx = fmaxf(mx, a);
  }
  float sum = 0.0f;
#pragma unroll 8
  for (int e = 0; e < DHEAD; ++e) {
    row[e] = expf(row[e] - mx);
    sum += row[e];
  }
  const float inv = 1.0f / sum;
#pragma unroll 8
  for (int e = 0; e < DHEAD; ++e)
    attn_s[head * DHEAD * DHEAD + e * DHEAD + d] = (_Float16)(row[e] * inv);
}

// ---------------------------------------------------------------------------
// Kernel 5: TDM loads the block's V-tile into LDS, then xo = attn @ V (WMMA,
// A-fragments from LDS), staged f16 to LDS, then out_c = xo @ Wproj + b.
// Block = 32 tokens, 8 waves.  LDS: v_s 16 KB + xo_s ~17 KB.
// ---------------------------------------------------------------------------
#define TOKS  32
#define XO_LD 264
__global__ __launch_bounds__(256) void k_xoproj(const _Float16* __restrict__ vv,
                                                const _Float16* __restrict__ attn_s,
                                                const float* __restrict__ wproj,
                                                const float* __restrict__ bproj,
                                                float* __restrict__ out) {
  __shared__ _Float16 v_s[TOKS * INNER];    // 16 KB, filled by TDM
  __shared__ _Float16 xo_s[TOKS * XO_LD];
  const long tok0 = (long)blockIdx.x * TOKS;
  const int  wave = threadIdx.x >> 5;
  const int  lane = threadIdx.x & 31;
  const int  nl   = lane & 15;
  const int  mb   = (lane & 16) ? 8 : 0;

  // --- Tensor Data Mover: async copy v[tok0 .. tok0+32) x 256ch f16 -> LDS.
  if (wave == 0) {                           // one TDM op per block
    const unsigned long long ga =
        (unsigned long long)(const void*)(vv + tok0 * INNER);
    const unsigned ldsa  = (unsigned)(unsigned long long)(void*)v_s;
    const unsigned elems = TOKS * INNER;     // 8192 x 2B = 16 KB, 1-D tile
    u32x4 g0;
    g0[0] = 1u;                                            // count=1
    g0[1] = ldsa;                                          // lds_addr
    g0[2] = (unsigned)(ga & 0xffffffffull);                // global_addr lo
    g0[3] = (unsigned)((ga >> 32) & 0x01ffffffull) | (2u << 30);  // hi | type=2
    i32x8 g1;
    g1[0] = (1 << 16);               // wg_mask=0, data_size=1 (2 bytes)
    g1[1] = (int)(elems << 16);      // tensor_dim0[15:0] in [31:16]
    g1[2] = (int)(1u << 16);         // tensor_dim0[31:16]=0 | tensor_dim1=1
    g1[3] = (int)(elems << 16);      // tile_dim0 in [31:16]
    g1[4] = 1;                       // tile_dim1=1, tile_dim2=0
    g1[5] = (int)elems;              // tensor_dim0_stride[31:0]
    g1[6] = 0;                       // stride hi | tensor_dim1_stride lo
    g1[7] = 0;
    i32x4 z4 = {0, 0, 0, 0};
    i32x8 z8 = {0, 0, 0, 0, 0, 0, 0, 0};
    // clang-23 / therock-10.0 form: 6 args (extra int32x8 group before cpol)
    __builtin_amdgcn_tensor_load_to_lds(g0, g1, z4, z4, z8, 0);
  }
  __builtin_amdgcn_s_wait_tensorcnt(0);      // s_wait_tensorcnt 0
  __syncthreads();

  // Step A: xo[tok, h*64+d] = sum_e v[tok, h*64+e] * attn[h][d][e]
  for (int t = wave * 4; t < wave * 4 + 4; ++t) {   // 2 row-tiles x 16 col-tiles
    const int rt = t >> 4, ct = t & 15;
    const int head = ct >> 2, dtl = ct & 3;
    v8f acc = {};
#pragma unroll
    for (int k0 = 0; k0 < DHEAD; k0 += 32) {
      v16h a = frag_a_f16(v_s + (rt * 16) * INNER + head * DHEAD + k0, INNER);
      v16h b = frag_b_f16(attn_s + head * DHEAD * DHEAD + (long)k0 * DHEAD +
                              dtl * 16,
                          DHEAD);
      acc = wmma16(a, b, acc);
    }
#pragma unroll
    for (int i = 0; i < 8; ++i)
      xo_s[(rt * 16 + mb + i) * XO_LD + ct * 16 + nl] = (_Float16)acc[i];
  }
  __syncthreads();

  // Step B: out[tok, c] = xo[tok, :] @ wproj[:, c] + b[c]
  {
    const int t  = wave;                              // 2 row-tiles x 4 col-tiles
    const int rt = t >> 2, ct = t & 3;
    v8f acc = {};
#pragma unroll
    for (int k0 = 0; k0 < INNER; k0 += 32) {
      v16h a = frag_a_f16(&xo_s[(rt * 16) * XO_LD + k0], XO_LD);  // ds_load_b128
      v16h b = frag_b_f32(wproj + (long)k0 * CDIM + ct * 16, CDIM);
      acc = wmma16(a, b, acc);
    }
    const float bias = bproj[ct * 16 + nl];
#pragma unroll
    for (int i = 0; i < 8; ++i)
      out[(tok0 + rt * 16 + mb + i) * CDIM + ct * 16 + nl] = acc[i] + bias;
  }
}

// ---------------------------------------------------------------------------
// Kernels 6/7: positional branch. Depthwise 3x3x3 conv, exact GELU, conv, add.
// ---------------------------------------------------------------------------
#define DD 8
#define HH 128
#define WW 128

__device__ inline float dwconv27(const float* __restrict__ src,
                                 const float* __restrict__ w27,
                                 int z, int y, int x, int c) {
  float acc = 0.0f;
#pragma unroll
  for (int dz = -1; dz <= 1; ++dz) {
    const int zz = z + dz;
    if (zz < 0 || zz >= DD) continue;
#pragma unroll
    for (int dy = -1; dy <= 1; ++dy) {
      const int yy = y + dy;
      if (yy < 0 || yy >= HH) continue;
#pragma unroll
      for (int dx = -1; dx <= 1; ++dx) {
        const int xx = x + dx;
        if (xx < 0 || xx >= WW) continue;
        const long sidx = ((long)zz * HH * WW + (long)yy * WW + xx) * CDIM + c;
        acc = fmaf(src[sidx], w27[(dz + 1) * 9 + (dy + 1) * 3 + (dx + 1)], acc);
      }
    }
  }
  return acc;
}

__global__ __launch_bounds__(256) void k_conv1_gelu(const float* __restrict__ x,
                                                    const float* __restrict__ wpe1,
                                                    float* __restrict__ t1) {
  const long tid = (long)blockIdx.x * blockDim.x + threadIdx.x;
  if (tid >= (long)NTOK * CDIM) return;
  const int  c = (int)(tid & (CDIM - 1));
  const long s = tid >> 6;
  const int  xx = (int)(s & (WW - 1));
  const int  yy = (int)((s >> 7) & (HH - 1));
  const int  zz = (int)(s >> 14);
  float v = dwconv27(x, wpe1 + c * 27, zz, yy, xx, c);
  v = 0.5f * v * (1.0f + erff(v * 0.70710678118654752f));  // exact GELU
  t1[tid] = v;
}

__global__ __launch_bounds__(256) void k_conv2_add(const float* __restrict__ t1,
                                                   const float* __restrict__ wpe2,
                                                   float* __restrict__ out) {
  const long tid = (long)blockIdx.x * blockDim.x + threadIdx.x;
  if (tid >= (long)NTOK * CDIM) return;
  const int  c = (int)(tid & (CDIM - 1));
  const long s = tid >> 6;
  const int  xx = (int)(s & (WW - 1));
  const int  yy = (int)((s >> 7) & (HH - 1));
  const int  zz = (int)(s >> 14);
  const float v = dwconv27(t1, wpe2 + c * 27, zz, yy, xx, c);
  out[tid] = out[tid] + v;   // out already holds attention branch
}

// ---------------------------------------------------------------------------
extern "C" void kernel_launch(void* const* d_in, const int* in_sizes, int n_in,
                              void* d_out, int out_size, void* d_ws, size_t ws_size,
                              hipStream_t stream) {
  const float* x_in    = (const float*)d_in[0];
  const float* w_qkv   = (const float*)d_in[1];
  const float* rescale = (const float*)d_in[2];
  const float* w_proj  = (const float*)d_in[3];
  const float* b_proj  = (const float*)d_in[4];
  const float* w_pe1   = (const float*)d_in[5];
  const float* w_pe2   = (const float*)d_in[6];
  float*       out     = (float*)d_out;

  // Workspace carve-up (t1 aliases qt, which is dead before the conv branch).
  const size_t QKB = (size_t)INNER * NTOK * sizeof(_Float16);  // 64 MiB each
  char* w = (char*)d_ws;
  _Float16* qt     = (_Float16*)(w);
  _Float16* kt     = (_Float16*)(w + QKB);
  _Float16* vv     = (_Float16*)(w + 2 * QKB);
  float*    sumsq  = (float*)(w + 3 * QKB);                       // 512 f32
  float*    G      = (float*)(w + 3 * QKB + 4096);                // 4*64*64 f32
  _Float16* attn_s = (_Float16*)(w + 3 * QKB + 4096 + 65536);     // 4*64*64 f16
  float*    t1     = (float*)(w);                                 // alias qt

  // 0) zero Gram accumulator
  k_zero_g<<<(HEADS * DHEAD * DHEAD + 255) / 256, 256, 0, stream>>>(G);

  // 1) QKV GEMM: (N/16)*(768/16) = 393216 wave-tiles, 8 waves/block
  {
    const long tiles = (long)(NTOK / 16) * (QKVW / 16);
    k_qkv<<<(int)(tiles / 8), 256, 0, stream>>>(x_in, w_qkv, qt, kt, vv);
  }

  // 2) per-channel sum of squares (512 channels)
  k_sumsq<<<2 * INNER, 256, 0, stream>>>(qt, kt, sumsq);

  // 3) Gram with split-K over N: 16 chunks of 8192
  {
    dim3 grid(HEADS, 16);
    k_gram<<<grid, 256, 0, stream>>>(qt, kt, G, NTOK / 16);
  }

  // 4) normalize + softmax -> attn^T (f16)
  k_attn<<<HEADS, 64, 0, stream>>>(G, sumsq, rescale, attn_s);

  // 5) TDM -> LDS -> WMMA: attn@V + projection
  k_xoproj<<<NTOK / TOKS, 256, 0, stream>>>(vv, attn_s, w_proj, b_proj, out);

  // 6/7) positional branch
  {
    const long elems = (long)NTOK * CDIM;
    const int  blocks = (int)((elems + 255) / 256);
    k_conv1_gelu<<<blocks, 256, 0, stream>>>(x_in, w_pe1, t1);
    k_conv2_add<<<blocks, 256, 0, stream>>>(t1, w_pe2, out);
  }
}
